// EncoderLSTM_16149077033020
// MI455X (gfx1250) — matrix operand order repaired and checked by reading, not compiled, not run on hardware
//
#include <hip/hip_runtime.h>
#include <hip/hip_bf16.h>
#include <math.h>

// ---------------- problem constants ----------------
constexpr int E_ = 512;      // embedding dim
constexpr int H_ = 1024;     // hidden
constexpr int D_ = 1024;     // head output dim
constexpr int B_ = 128;      // batch
constexpr int T_ = 256;      // time
constexpr int G_ = 4 * H_;   // 4096 gates
constexpr int KC_ = E_ + H_; // 1536 fused K (input proj + recurrent)

typedef __attribute__((ext_vector_type(16))) _Float16 v16h;
typedef __attribute__((ext_vector_type(8)))  _Float16 v8h;
typedef __attribute__((ext_vector_type(8)))  float    v8f;
typedef __attribute__((ext_vector_type(4)))  int      v4i;

__device__ __forceinline__ float sigm(float x) { return 1.0f / (1.0f + expf(-x)); }

// ---------------- CDNA5 async copy global->LDS -----
#if __has_builtin(__builtin_amdgcn_global_load_async_to_lds_b128)
#define HAVE_ASYNC_LDS 1
typedef __attribute__((address_space(1))) v4i* as1_v4i;
typedef __attribute__((address_space(3))) v4i* as3_v4i;
#else
#define HAVE_ASYNC_LDS 0
#endif

__device__ __forceinline__ void cp16(_Float16* ldsDst, const _Float16* gsrc) {
#if HAVE_ASYNC_LDS
  // memory -> LDS DMA, tracked by ASYNCcnt (no VGPR round trip)
  __builtin_amdgcn_global_load_async_to_lds_b128(
      (as1_v4i)(v4i*)gsrc,     // global source (addrspace 1)
      (as3_v4i)(v4i*)ldsDst,   // LDS destination (addrspace 3)
      /*offset=*/0, /*cpol=*/0);
#else
  *(uint4*)ldsDst = *(const uint4*)gsrc;
#endif
}

__device__ __forceinline__ void async_join() {
#if HAVE_ASYNC_LDS
#if __has_builtin(__builtin_amdgcn_s_wait_asynccnt)
  __builtin_amdgcn_s_wait_asynccnt(0);
#else
  asm volatile("s_wait_asynccnt 0x0" ::: "memory");
#endif
#endif
}

// ---------------- WMMA tile helper -----------------
// 64x64 C tile per workgroup, 4 waves, each wave owns 16 rows x 64 cols
// (4 accumulators of 16x16). A/B staged in LDS as [64 rows][32 k] f16.
// A frag (16-bit 16x32): lanes 0-15 hold K {0..7,16..23}; lanes 16-31 hold
// K {8..15,24..31} -> two 16B contiguous LDS reads per lane.
// B frag (32x16 KxN): lane&15 = N, same K split as A (lanes 0-15 K {0..7,16..23}).
union U16 { v16h v; v8h h[2]; };

__device__ __forceinline__ void wmma_tile_accum(const _Float16* As, const _Float16* Bs,
                                                int wave, int lane, v8f (&acc)[4]) {
  const int half = lane >> 4;
  const int arow = wave * 16 + (lane & 15);
  U16 a;
  a.h[0] = *(const v8h*)(As + arow * 32 + half * 8);
  a.h[1] = *(const v8h*)(As + arow * 32 + 16 + half * 8);
#pragma unroll
  for (int s = 0; s < 4; ++s) {
    const int brow = s * 16 + (lane & 15);
    U16 b;
    b.h[0] = *(const v8h*)(Bs + brow * 32 + half * 8);
    b.h[1] = *(const v8h*)(Bs + brow * 32 + 16 + half * 8);
    acc[s] = __builtin_amdgcn_wmma_f32_16x16x32_f16(
        /*neg_a=*/false, a.v, /*neg_b=*/false, b.v,
        /*c_mod=*/(short)0, acc[s], /*reuse_a=*/false, /*reuse_b=*/false);
  }
}

// Double-buffered K loop over one 64x64 output tile.
// lds layout: buf0 A(2048h) B(2048h) | buf1 A(2048h) B(2048h)  -> 16 KB
template <typename ASrc>
__device__ __forceinline__ void run_gemm_tile(ASrc asrc,            // (row, gk) -> const _Float16*
                                              const _Float16* wbase,
                                              size_t wstride, int n0, int K,
                                              _Float16* lds, int tid,
                                              v8f (&acc)[4]) {
  const int wave = tid >> 5, lane = tid & 31;
  const int row = tid >> 1;        // 0..63
  const int c0 = (tid & 1) * 16;   // 0 or 16 halves

  auto stage = [&](int k0, int buf) {
    const int gk = k0 + c0;
    _Float16* base = lds + buf * 4096;
    const _Float16* ap = asrc(row, gk);
    cp16(base + row * 32 + c0, ap);
    cp16(base + row * 32 + c0 + 8, ap + 8);
    const _Float16* wp = wbase + (size_t)(n0 + row) * wstride + gk;
    cp16(base + 2048 + row * 32 + c0, wp);
    cp16(base + 2048 + row * 32 + c0 + 8, wp + 8);
    __builtin_prefetch(wp + 32, 0, 3);  // L2 hint for next weight chunk
  };

  const int NK = K / 32;
  stage(0, 0);
  for (int kc = 0; kc < NK; ++kc) {
    async_join();       // this wave's async LDS writes landed
    __syncthreads();    // everyone's writes landed; prev reads retired
    if (kc + 1 < NK) stage((kc + 1) * 32, (kc + 1) & 1);  // overlap with WMMA
    const _Float16* base = lds + (kc & 1) * 4096;
    wmma_tile_accum(base, base + 2048, wave, lane, acc);
  }
}

// ---------------- prep kernels ---------------------
__global__ void k_embed(const long long* __restrict__ tok,
                        const float* __restrict__ emb,
                        _Float16* __restrict__ xh) {
  size_t idx = (size_t)blockIdx.x * blockDim.x + threadIdx.x; // over T*B*E
  int e = idx % E_;
  int b = (int)((idx / E_) % B_);
  int t = (int)(idx / ((size_t)E_ * B_));
  long long v = tok[(size_t)b * T_ + t];
  float val = (v == 0) ? 0.0f : emb[(size_t)v * E_ + e]; // PAD==0 row zeroed
  xh[idx] = (_Float16)val;
}

__global__ void k_prep_wcat(const float* __restrict__ Wih,
                            const float* __restrict__ Whh,
                            _Float16* __restrict__ wcat) {
  size_t idx = (size_t)blockIdx.x * blockDim.x + threadIdx.x; // over G_*KC_
  if (idx >= (size_t)G_ * KC_) return;
  int k = (int)(idx % KC_);
  int n = (int)(idx / KC_);
  float v = (k < E_) ? Wih[(size_t)n * E_ + k] : Whh[(size_t)n * H_ + (k - E_)];
  wcat[idx] = (_Float16)v;
}

__global__ void k_bias_sum(const float* a, const float* b, float* o, int n) {
  int i = blockIdx.x * blockDim.x + threadIdx.x;
  if (i < n) o[i] = a[i] + b[i];
}

__global__ void k_cvt_f16(const float* __restrict__ s, _Float16* __restrict__ d, size_t n) {
  size_t i = (size_t)blockIdx.x * blockDim.x + threadIdx.x;
  if (i < n) d[i] = (_Float16)s[i];
}

__global__ void k_zero_f32(float* p, size_t n) {
  size_t i = (size_t)blockIdx.x * blockDim.x + threadIdx.x;
  if (i < n) p[i] = 0.0f;
}
__global__ void k_zero_f16(_Float16* p, size_t n) {
  size_t i = (size_t)blockIdx.x * blockDim.x + threadIdx.x;
  if (i < n) p[i] = (_Float16)0.0f;
}

// ---------------- recurrent step: gate GEMM --------
// gates[dir][b][n] = sum_k [x_t | h] * wcat[dir][n][k] + bias[dir][n]
// grid: (G_/64, B_/64, 2dirs), block 128 (4 waves), 16 KB dynamic LDS
__global__ void k_gate_gemm(const _Float16* __restrict__ xh,   // (T,B,E)
                            const _Float16* __restrict__ h16,  // (2,B,H)
                            const _Float16* __restrict__ wcat, // (2,G,KC)
                            const float* __restrict__ bias,    // (2,G)
                            float* __restrict__ gates,         // (2,B,G)
                            int t) {
  extern __shared__ _Float16 lds[];
  const int dir = blockIdx.z;
  const int t_eff = dir ? (T_ - 1 - t) : t;
  const int n0 = blockIdx.x * 64;
  const int m0 = blockIdx.y * 64;
  const int tid = threadIdx.x;
  const int wave = tid >> 5, lane = tid & 31;

  const _Float16* xrow = xh + (size_t)t_eff * B_ * E_;
  const _Float16* hrow = h16 + (size_t)dir * B_ * H_;
  const _Float16* wrow = wcat + (size_t)dir * G_ * KC_;

  v8f acc[4] = {};
  auto asrc = [&](int r, int gk) -> const _Float16* {
    return (gk < E_) ? xrow + (size_t)(m0 + r) * E_ + gk
                     : hrow + (size_t)(m0 + r) * H_ + (gk - E_);
  };
  run_gemm_tile(asrc, wrow, (size_t)KC_, n0, KC_, lds, tid, acc);

  const int half = lane >> 4;
  const int ncol = lane & 15;
  const int mrow = m0 + wave * 16 + half * 8;
  float* gbase = gates + (size_t)dir * B_ * G_;
#pragma unroll
  for (int s = 0; s < 4; ++s) {
    const int n = n0 + s * 16 + ncol;
    const float bv = bias[dir * G_ + n];
#pragma unroll
    for (int r = 0; r < 8; ++r)
      gbase[(size_t)(mrow + r) * G_ + n] = acc[s][r] + bv;
  }
}

// ---------------- recurrent step: cell -------------
__global__ void k_cell(const float* __restrict__ gates,     // (2,B,G)
                       const long long* __restrict__ lens,  // (B)
                       float* __restrict__ h_state,         // (2,B,H)
                       float* __restrict__ c_state,         // (2,B,H)
                       _Float16* __restrict__ h16,          // (2,B,H)
                       float* __restrict__ ctx,             // (B,T,2H) in d_out
                       int t) {
  int idx = blockIdx.x * blockDim.x + threadIdx.x; // over 2*B*H
  if (idx >= 2 * B_ * H_) return;
  const int j = idx % H_;
  const int b = (idx / H_) % B_;
  const int dir = idx / (B_ * H_);
  const int t_eff = dir ? (T_ - 1 - t) : t;
  const bool m = t_eff < (int)lens[b];

  const float* g = gates + ((size_t)dir * B_ + b) * G_;
  const float gi = g[j], gf = g[H_ + j], gg = g[2 * H_ + j], go = g[3 * H_ + j];

  const size_t s = ((size_t)dir * B_ + b) * H_ + j;
  const float c_old = c_state[s], h_old = h_state[s];
  const float c_new = sigm(gf) * c_old + sigm(gi) * tanhf(gg);
  const float h_new = sigm(go) * tanhf(c_new);
  const float h_o = m ? h_new : h_old;
  const float c_o = m ? c_new : c_old;
  h_state[s] = h_o;
  c_state[s] = c_o;
  h16[s] = (_Float16)h_o;
  ctx[((size_t)b * T_ + t_eff) * (2 * H_) + dir * H_ + j] = m ? h_new : 0.0f;
}

// ---------------- head: concat + GEMM --------------
__global__ void k_build_cat(const float* __restrict__ h_state,
                            const float* __restrict__ c_state,
                            _Float16* __restrict__ hcat,   // (B, 2H), backward first
                            _Float16* __restrict__ ccat) {
  int idx = blockIdx.x * blockDim.x + threadIdx.x; // over B*2H
  if (idx >= B_ * 2 * H_) return;
  const int j = idx % (2 * H_);
  const int b = idx / (2 * H_);
  const int dir = (j < H_) ? 1 : 0;     // h_t = concat([h_b, h_f])
  const int jj = (j < H_) ? j : j - H_;
  const size_t s = ((size_t)dir * B_ + b) * H_ + jj;
  hcat[idx] = (_Float16)h_state[s];
  ccat[idx] = (_Float16)c_state[s];
}

// grid: (D_/64, B_/64, 2)  z=0 -> h (tanh), z=1 -> c (linear); 16 KB LDS
__global__ void k_head_gemm(const _Float16* __restrict__ hcat,
                            const _Float16* __restrict__ ccat,
                            const _Float16* __restrict__ w2h,  // (D, 2H) f16
                            const _Float16* __restrict__ w2c,
                            const float* __restrict__ b2h,
                            const float* __restrict__ b2c,
                            float* __restrict__ outh,
                            float* __restrict__ outc) {
  extern __shared__ _Float16 lds[];
  constexpr int K = 2 * H_;
  const int z = blockIdx.z;
  const _Float16* A = z ? ccat : hcat;
  const _Float16* W = z ? w2c : w2h;
  const float* bias = z ? b2c : b2h;
  float* out = z ? outc : outh;

  const int n0 = blockIdx.x * 64;
  const int m0 = blockIdx.y * 64;
  const int tid = threadIdx.x;
  const int wave = tid >> 5, lane = tid & 31;

  v8f acc[4] = {};
  auto asrc = [&](int r, int gk) -> const _Float16* {
    return A + (size_t)(m0 + r) * K + gk;
  };
  run_gemm_tile(asrc, W, (size_t)K, n0, K, lds, tid, acc);

  const int half = lane >> 4;
  const int ncol = lane & 15;
  const int mrow = m0 + wave * 16 + half * 8;
#pragma unroll
  for (int s = 0; s < 4; ++s) {
    const int n = n0 + s * 16 + ncol;
#pragma unroll
    for (int r = 0; r < 8; ++r) {
      float v = acc[s][r] + bias[n];
      if (!z) v = tanhf(v);
      out[(size_t)(mrow + r) * D_ + n] = v;
    }
  }
}

// ---------------- host side ------------------------
static inline size_t alignUp(size_t x) { return (x + 255) & ~(size_t)255; }

extern "C" void kernel_launch(void* const* d_in, const int* in_sizes, int n_in,
                              void* d_out, int out_size, void* d_ws, size_t ws_size,
                              hipStream_t stream) {
  (void)in_sizes; (void)n_in; (void)out_size; (void)ws_size;

  const long long* tok   = (const long long*)d_in[0];
  const long long* lens  = (const long long*)d_in[1];
  const float* emb       = (const float*)d_in[2];
  const float* Wih_f     = (const float*)d_in[3];
  const float* Whh_f     = (const float*)d_in[4];
  const float* bih_f     = (const float*)d_in[5];
  const float* bhh_f     = (const float*)d_in[6];
  const float* Wih_b     = (const float*)d_in[7];
  const float* Whh_b     = (const float*)d_in[8];
  const float* bih_b     = (const float*)d_in[9];
  const float* bhh_b     = (const float*)d_in[10];
  const float* W2h_w     = (const float*)d_in[11];
  const float* W2h_b     = (const float*)d_in[12];
  const float* W2c_w     = (const float*)d_in[13];
  const float* W2c_b     = (const float*)d_in[14];

  float* out       = (float*)d_out;
  float* ctx       = out;                                   // (B,T,2H)
  float* out_h     = out + (size_t)B_ * T_ * 2 * H_;        // (B,D)
  float* out_c     = out_h + (size_t)B_ * D_;               // (B,D)

  // -------- workspace carve-out --------
  char* ws = (char*)d_ws;
  size_t off = 0;
  auto take = [&](size_t bytes) { char* p = ws + off; off = alignUp(off + bytes); return p; };
  _Float16* xh      = (_Float16*)take((size_t)T_ * B_ * E_ * 2);       // 32 MB
  _Float16* wcat    = (_Float16*)take((size_t)2 * G_ * KC_ * 2);       // 24 MB
  float*    bias    = (float*)   take((size_t)2 * G_ * 4);
  _Float16* w2h16   = (_Float16*)take((size_t)D_ * 2 * H_ * 2);        // 4 MB
  _Float16* w2c16   = (_Float16*)take((size_t)D_ * 2 * H_ * 2);        // 4 MB
  float*    h_state = (float*)   take((size_t)2 * B_ * H_ * 4);
  float*    c_state = (float*)   take((size_t)2 * B_ * H_ * 4);
  _Float16* h16     = (_Float16*)take((size_t)2 * B_ * H_ * 2);
  float*    gates   = (float*)   take((size_t)2 * B_ * G_ * 4);        // 4 MB
  _Float16* hcat    = (_Float16*)take((size_t)B_ * 2 * H_ * 2);
  _Float16* ccat    = (_Float16*)take((size_t)B_ * 2 * H_ * 2);

  // -------- one-time (per call) prep --------
  {
    size_t n = (size_t)T_ * B_ * E_;
    k_embed<<<dim3((unsigned)(n / 256)), 256, 0, stream>>>(tok, emb, xh);
  }
  {
    size_t n = (size_t)G_ * KC_;
    unsigned blk = (unsigned)((n + 255) / 256);
    k_prep_wcat<<<blk, 256, 0, stream>>>(Wih_f, Whh_f, wcat);
    k_prep_wcat<<<blk, 256, 0, stream>>>(Wih_b, Whh_b, wcat + n);
  }
  k_bias_sum<<<(G_ + 255) / 256, 256, 0, stream>>>(bih_f, bhh_f, bias, G_);
  k_bias_sum<<<(G_ + 255) / 256, 256, 0, stream>>>(bih_b, bhh_b, bias + G_, G_);
  {
    size_t n = (size_t)D_ * 2 * H_;
    unsigned blk = (unsigned)((n + 255) / 256);
    k_cvt_f16<<<blk, 256, 0, stream>>>(W2h_w, w2h16, n);
    k_cvt_f16<<<blk, 256, 0, stream>>>(W2c_w, w2c16, n);
  }
  {
    size_t n = (size_t)2 * B_ * H_;
    unsigned blk = (unsigned)((n + 255) / 256);
    k_zero_f32<<<blk, 256, 0, stream>>>(h_state, n);
    k_zero_f32<<<blk, 256, 0, stream>>>(c_state, n);
    k_zero_f16<<<blk, 256, 0, stream>>>(h16, n);
  }

  // -------- recurrent scan (both directions per step) --------
  const dim3 ggrid(G_ / 64, B_ / 64, 2);   // (64, 2, 2)
  const unsigned cblk = (2 * B_ * H_) / 256;
  const size_t ldsBytes = 16384;           // double-buffered A+B tiles
  for (int t = 0; t < T_; ++t) {
    k_gate_gemm<<<ggrid, 128, ldsBytes, stream>>>(xh, h16, wcat, bias, gates, t);
    k_cell<<<cblk, 256, 0, stream>>>(gates, lens, h_state, c_state, h16, ctx, t);
  }

  // -------- heads --------
  k_build_cat<<<(B_ * 2 * H_) / 256, 256, 0, stream>>>(h_state, c_state, hcat, ccat);
  const dim3 hgrid(D_ / 64, B_ / 64, 2);   // (16, 2, 2)
  k_head_gemm<<<hgrid, 128, ldsBytes, stream>>>(hcat, ccat, w2h16, w2c16, W2h_b, W2c_b, out_h, out_c);
}
